// SoftTokenAlignerBridge_88819923682052
// MI455X (gfx1250) — matrix operand order repaired and checked by reading, not compiled, not run on hardware
//
#include <hip/hip_runtime.h>
#include <hip/hip_bf16.h>
#include <stdint.h>

typedef __attribute__((ext_vector_type(16))) _Float16     v16h;
typedef __attribute__((ext_vector_type(8)))  _Float16     v8h;
typedef __attribute__((ext_vector_type(8)))  float        v8f;
typedef __attribute__((ext_vector_type(8)))  unsigned int v8u;

#define Bn      16
#define Hn      1024
#define Tn      1500
#define TpN     1504      // padded T (multiple of 32)
#define MT      192       // MAX_TOK
#define KTILES  12        // MT/16
#define HTILES  64        // Hn/16
#define TTILES  94        // ceil(Tn/16)
#define TCH     47        // TpN/32 (K chunks for final GEMM)
#define ICH     32        // Hn/32 (K chunks for conv GEMM)
#define INV2T2  4.08163265306f   // 1/(2*0.35^2)
#define EPSF    1e-6f

__device__ __forceinline__ unsigned int pack2h(float lo, float hi) {
    union { _Float16 f; unsigned short u; } a, b;
    a.f = (_Float16)lo; b.f = (_Float16)hi;
    return ((unsigned int)b.u << 16) | (unsigned int)a.u;
}

// ---------------------------------------------------------------- init loss
__global__ void init_loss_kernel(float* loss) { *loss = 0.f; }

// ------------------------------------------------- f32 -> f16 padded x copy
// xh[b][h][t] (Tp padded with zeros); also the A operand of the final GEMM.
__global__ __launch_bounds__(256) void prep_xh_kernel(const float* __restrict__ enc,
                                                      _Float16* __restrict__ xh) {
    size_t idx = (size_t)blockIdx.x * 256 + threadIdx.x;   // B*Hn*TpN total
    int t = (int)(idx % TpN);
    size_t row = idx / TpN;                                 // b*Hn + h
    float v = (t < Tn) ? enc[row * (size_t)Tn + t] : 0.f;
    xh[idx] = (_Float16)v;
}

// ----------------------------------- conv weights -> WMMA B-fragment order
// wfrag[k][ic][ht][lane][j] : dword packs rows {r, r+1} of B (K=i dim), col = h.
__global__ __launch_bounds__(256) void prep_wfrag_kernel(const float* __restrict__ cw,
                                                         unsigned int* __restrict__ wfrag) {
    unsigned int idx = blockIdx.x * 256 + threadIdx.x;      // 3*ICH*HTILES*32*8 total
    int jd   = idx & 7;
    int lane = (idx >> 3) & 31;
    int ht   = (idx >> 8) & 63;
    int ic   = (idx >> 14) & 31;
    int k    = idx >> 19;
    int h = ht * 16 + (lane & 15);
    int i = ic * 32 + ((lane < 16) ? 0 : 8) + (jd >> 2) * 16 + (jd & 3) * 2;
    // conv_w layout [O=h][I=i][K=k]
    float lo = cw[((size_t)h * Hn + i) * 3 + k];
    float hi = cw[((size_t)h * Hn + (i + 1)) * 3 + k];
    wfrag[idx] = pack2h(lo, hi);
}

// --------------------------------------- conv(GEMM) + silu + proj + softplus
// One block per (t-tile, b); 8 waves, each owning 8 h-tiles. Produces raw[B,Tp].
__global__ __launch_bounds__(256) void conv_raw_kernel(const _Float16* __restrict__ xh,
                                                       const unsigned int* __restrict__ wfrag,
                                                       const float* __restrict__ conv_b,
                                                       const float* __restrict__ proj_w,
                                                       const float* __restrict__ proj_b,
                                                       const int* __restrict__ enc_len,
                                                       float* __restrict__ raw) {
    int b  = blockIdx.y;
    int t0 = blockIdx.x * 16;
    int tid  = threadIdx.x;
    int wave = tid >> 5;
    int lane = tid & 31;

    // Transposed x tile: At[t-local(0..17)][i-local(0..31)] so each lane's
    // A-fragment is two contiguous 16B chunks -> 2 x ds_load_b128.
    __shared__ __align__(16) _Float16 At[18 * 32];
    __shared__ float rawacc[16];
    if (tid < 16) rawacc[tid] = 0.f;

    const v8f vzero = {0.f,0.f,0.f,0.f,0.f,0.f,0.f,0.f};
    v8f acc[8];
#pragma unroll
    for (int j = 0; j < 8; ++j) acc[j] = vzero;

    const int m     = lane & 15;            // A-matrix row (t)
    const int kbase = (lane < 16) ? 0 : 8;  // A-matrix K group offset
    const _Float16* xb = xh + (size_t)b * Hn * TpN;
    // Per-lane LDS fragment base (halves): row m, K-group kbase
    const _Float16* afrag = At + m * 32 + kbase;

    union BU { v8u u; v16h h; };
    union AU { v8h p[2]; v16h h; };

    for (int ic = 0; ic < ICH; ++ic) {
        __syncthreads();
        // cooperative stage of x tile (halo for 3-tap SAME conv)
        for (int idx = tid; idx < 32 * 18; idx += 256) {
            int i  = idx / 18;
            int tl = idx % 18;
            int tg = t0 - 1 + tl;
            _Float16 v = (_Float16)0.f;
            if (tg >= 0 && tg < Tn) v = xb[(size_t)(ic * 32 + i) * TpN + tg];
            At[tl * 32 + i] = v;
        }
        __syncthreads();
#pragma unroll
        for (int k = 0; k < 3; ++k) {
            // A fragment: 2 contiguous 16B LDS loads (rows m+k, K groups)
            AU am;
            am.p[0] = *(const v8h*)(afrag + k * 32);        // K = kbase+0..7
            am.p[1] = *(const v8h*)(afrag + k * 32 + 16);   // K = kbase+16..23
            // Batch-preload all 8 B fragments (constant 1KB offsets per j)
            const unsigned int* wp =
                wfrag + ((((size_t)k * ICH + ic) * HTILES + wave * 8) * 32 + lane) * 8;
            BU bm[8];
#pragma unroll
            for (int j = 0; j < 8; ++j) bm[j].u = *(const v8u*)(wp + j * 256);
#pragma unroll
            for (int j = 0; j < 8; ++j)
                acc[j] = __builtin_amdgcn_wmma_f32_16x16x32_f16(
                            false, am.h, false, bm[j].h, (short)0, acc[j], false, false);
        }
    }

    // Epilogue: bias -> silu -> * proj_w, reduce across h into rawacc[t-row]
    const int ncol = lane & 15;
    const int moff = (lane < 16) ? 0 : 8;
#pragma unroll
    for (int j = 0; j < 8; ++j) {
        int h = (wave * 8 + j) * 16 + ncol;
        float cb = conv_b[h];
        float pw = proj_w[h];
#pragma unroll
        for (int r = 0; r < 8; ++r) {
            float x = acc[j][r] + cb;
            float s = x / (1.f + __expf(-x));          // silu
            atomicAdd(&rawacc[moff + r], s * pw);      // ds_add_f32
        }
    }
    __syncthreads();
    if (tid < 16) {
        int t = t0 + tid;
        float x  = rawacc[tid] + proj_b[0];
        float sp = (x > 20.f) ? x : log1pf(__expf(x));
        float v  = sp + 1e-4f;
        if (t >= enc_len[b] || t >= Tn) v = 0.f;       // time mask / padding
        raw[(size_t)b * TpN + t] = v;
    }
}

// -------------------- per-batch: mass, loss, tok_len, scale, scan -> centers
__global__ __launch_bounds__(256) void scan_kernel(const float* __restrict__ raw,
                                                   const int* __restrict__ tgt,
                                                   float* __restrict__ sw,
                                                   float* __restrict__ centers,
                                                   float* __restrict__ tok_out,
                                                   float* __restrict__ loss) {
    int b = blockIdx.x, tid = threadIdx.x;
    __shared__ float ls[256];
    const float* rb = raw + (size_t)b * TpN;
    float* swb = sw + (size_t)b * TpN;
    float* cb  = centers + (size_t)b * TpN;

    float s = 0.f;
    for (int t = tid; t < TpN; t += 256) s += rb[t];
    ls[tid] = s; __syncthreads();
    for (int o = 128; o > 0; o >>= 1) { if (tid < o) ls[tid] += ls[tid + o]; __syncthreads(); }
    float mass = ls[0];

    int tok = tgt[b]; tok = (tok < 1) ? 1 : ((tok > MT) ? MT : tok);
    float scale = (float)tok / fmaxf(mass, EPSF);
    if (tid == 0) {
        tok_out[b] = (float)tok;
        atomicAdd(loss, fabsf(mass - (float)tok) * (0.25f / (float)Bn));
    }

    float carry = 0.f;
    for (int c = 0; c < 6; ++c) {                 // 6*256 >= TpN
        int t = c * 256 + tid;
        float v = (t < TpN) ? rb[t] * scale : 0.f;
        __syncthreads();
        ls[tid] = v; __syncthreads();
        for (int o = 1; o < 256; o <<= 1) {       // inclusive Hillis-Steele
            float add = (tid >= o) ? ls[tid - o] : 0.f;
            __syncthreads();
            ls[tid] += add;
            __syncthreads();
        }
        float incl = ls[tid];
        float tot  = ls[255];
        if (t < TpN) { swb[t] = v; cb[t] = carry + incl - 0.5f * v; }
        carry += tot;
    }
}

// --------------------------------------- per-(b,k) assignment normalizer
__global__ __launch_bounds__(256) void denom_kernel(const float* __restrict__ sw,
                                                    const float* __restrict__ centers,
                                                    float* __restrict__ denom) {
    int k = blockIdx.x, b = blockIdx.y, tid = threadIdx.x;
    __shared__ float ls[256];
    const float* swb = sw + (size_t)b * TpN;
    const float* cb  = centers + (size_t)b * TpN;
    float ctr = (float)k + 0.5f;
    float s = 0.f;
    for (int t = tid; t < TpN; t += 256) {
        float d = cb[t] - ctr;
        s += __expf(-d * d * INV2T2) * swb[t];
    }
    ls[tid] = s; __syncthreads();
    for (int o = 128; o > 0; o >>= 1) { if (tid < o) ls[tid] += ls[tid + o]; __syncthreads(); }
    if (tid == 0) denom[b * MT + k] = ls[0];
}

// ------------- build assignment matrix directly in WMMA B-fragment order
// bfrag[b][ktile][tc][lane][j] : dword packs assign rows {t, t+1}, col = k.
__global__ __launch_bounds__(256) void build_bfrag_kernel(const float* __restrict__ sw,
                                                          const float* __restrict__ centers,
                                                          const float* __restrict__ denom,
                                                          const int* __restrict__ tgt,
                                                          unsigned int* __restrict__ bfrag) {
    size_t idx = (size_t)blockIdx.x * 256 + threadIdx.x;   // Bn*KTILES*TCH*32*8 total
    int jd    = (int)(idx & 7);
    int lane  = (int)((idx >> 3) & 31);
    size_t q  = idx >> 8;
    int tc    = (int)(q % TCH); q /= TCH;
    int ktile = (int)(q % KTILES);
    int b     = (int)(q / KTILES);

    int k = ktile * 16 + (lane & 15);
    int tok = tgt[b]; tok = (tok < 1) ? 1 : ((tok > MT) ? MT : tok);
    float den = fmaxf(denom[b * MT + k], EPSF);
    float ctr = (float)k + 0.5f;
    int rbase = ((lane < 16) ? 0 : 8) + (jd >> 2) * 16 + (jd & 3) * 2;

    float v[2];
#pragma unroll
    for (int p = 0; p < 2; ++p) {
        int t = tc * 32 + rbase + p;
        float a = 0.f;
        if (t < Tn && k < tok) {
            float d = centers[(size_t)b * TpN + t] - ctr;
            a = __expf(-d * d * INV2T2) * sw[(size_t)b * TpN + t] / den;
        }
        v[p] = a;
    }
    bfrag[idx] = pack2h(v[0], v[1]);
}

// ----------------------- final einsum: acoustic[b,k,h] = sum_t assign * feat
// Computed as D[h,k] = A(feat[h,t]) x B(assign[t,k]); D layout stores 8
// contiguous floats of acoustic[b][k][h...] per lane. Fully unrolled so all
// loads use immediate offsets and can be hoisted across WMMAs.
__global__ __launch_bounds__(256) void token_gemm_kernel(const _Float16* __restrict__ xh,
                                                         const unsigned int* __restrict__ bfrag,
                                                         float* __restrict__ out) {
    int ktile = blockIdx.x;            // 0..11
    int hgrp  = blockIdx.y;            // 0..7
    int b     = blockIdx.z;
    int tid = threadIdx.x, wave = tid >> 5, lane = tid & 31;

    int h0   = (hgrp * 8 + wave) * 16;
    int hrow = h0 + (lane & 15);                 // A-matrix row (M)
    int tsel = (lane < 16) ? 0 : 8;              // A-matrix K group offset
    const _Float16* xr = xh + ((size_t)b * Hn + hrow) * TpN + tsel;
    const unsigned int* bp = bfrag + (((size_t)b * KTILES + ktile) * TCH * 32 + lane) * 8;

    union BU { v8u u; v16h h; };
    union AU { v8h p[2]; v16h h; };

    v8f acc = {0.f,0.f,0.f,0.f,0.f,0.f,0.f,0.f};
#pragma unroll
    for (int tc = 0; tc < TCH; ++tc) {
        AU am;
        am.p[0] = *(const v8h*)(xr + tc * 32);        // K = tsel+0..7
        am.p[1] = *(const v8h*)(xr + tc * 32 + 16);   // K = tsel+16..23
        BU bm;
        bm.u = *(const v8u*)(bp + tc * 256);
        acc = __builtin_amdgcn_wmma_f32_16x16x32_f16(
                  false, am.h, false, bm.h, (short)0, acc, false, false);
    }
    int k  = ktile * 16 + (lane & 15);           // D column (N)
    int hb = h0 + ((lane < 16) ? 0 : 8);         // D rows M = r (+8)
    float* op = out + ((size_t)b * MT + k) * Hn + hb;
#pragma unroll
    for (int r = 0; r < 8; ++r) op[r] = acc[r];
}

// ---------------------------------------------------------------------------
extern "C" void kernel_launch(void* const* d_in, const int* in_sizes, int n_in,
                              void* d_out, int out_size, void* d_ws, size_t ws_size,
                              hipStream_t stream) {
    const float* enc  = (const float*)d_in[0];
    const int*   elen = (const int*)  d_in[1];
    const int*   tlen = (const int*)  d_in[2];
    const float* cw   = (const float*)d_in[3];
    const float* cb   = (const float*)d_in[4];
    const float* pw   = (const float*)d_in[5];
    const float* pb   = (const float*)d_in[6];

    char* ws = (char*)d_ws;
    size_t off = 0;
    auto take = [&](size_t bytes) -> char* {
        char* p = ws + off;
        off = (off + bytes + 255) & ~(size_t)255;
        return p;
    };
    _Float16*     xh      = (_Float16*)    take((size_t)Bn * Hn * TpN * 2);
    unsigned int* wfrag   = (unsigned int*)take((size_t)3 * ICH * HTILES * 32 * 8 * 4);
    unsigned int* bfrag   = (unsigned int*)take((size_t)Bn * KTILES * TCH * 32 * 8 * 4);
    float*        raw     = (float*)       take((size_t)Bn * TpN * 4);
    float*        sw      = (float*)       take((size_t)Bn * TpN * 4);
    float*        centers = (float*)       take((size_t)Bn * TpN * 4);
    float*        denom   = (float*)       take((size_t)Bn * MT * 4);

    float* ac      = (float*)d_out;                    // [B, MT, H]
    float* tok_out = ac + (size_t)Bn * MT * Hn;        // [B]
    float* loss    = tok_out + Bn;                     // [1]

    init_loss_kernel<<<1, 1, 0, stream>>>(loss);
    prep_xh_kernel<<<(Bn * Hn * TpN) / 256, 256, 0, stream>>>(enc, xh);
    prep_wfrag_kernel<<<(3 * ICH * HTILES * 32 * 8) / 256, 256, 0, stream>>>(cw, wfrag);
    conv_raw_kernel<<<dim3(TTILES, Bn), 256, 0, stream>>>(xh, wfrag, cb, pw, pb, elen, raw);
    scan_kernel<<<Bn, 256, 0, stream>>>(raw, tlen, sw, centers, tok_out, loss);
    denom_kernel<<<dim3(MT, Bn), 256, 0, stream>>>(sw, centers, denom);
    build_bfrag_kernel<<<(Bn * KTILES * TCH * 32 * 8) / 256, 256, 0, stream>>>(sw, centers, denom, tlen, bfrag);
    token_gemm_kernel<<<dim3(KTILES, 8, Bn), 256, 0, stream>>>(xh, bfrag, ac);
}
